// DeformableConvV2_71116068487487
// MI455X (gfx1250) — matrix operand order repaired and checked by reading, not compiled, not run on hardware
//
#include <hip/hip_runtime.h>
#include <hip/hip_bf16.h>

typedef __attribute__((ext_vector_type(16))) _Float16 v16h;
typedef __attribute__((ext_vector_type(8)))  float    v8f;

#define Bn 4
#define CIN 256
#define HH 64
#define WW 64
#define COUT 256
#define KK 9
#define HW (HH * WW)          // 4096
#define NCHUNK 8              // 256 / 32 K-chunks per tap
#define NMT 16                // 256 / 16 M tiles
#define FRAG_HALVES 512       // 32 lanes * 16 halves

// ---------------------------------------------------------------------------
// Kernel 1: pack w_dcn (C_OUT, C_IN, 3, 3) f32 -> f16 A-fragments laid out as
// [kk][ci][mt][lane][j] matching the ISA 16-bit A-matrix 16x32 layout:
//   lane = hi*16 + m ;  element j -> K = (v<4 ? 2v : 16+2(v-4)) + pos + 8*hi
// ---------------------------------------------------------------------------
__global__ void dcn_pack_wdcn(const float* __restrict__ w_dcn,
                              _Float16* __restrict__ apack) {
    int id = blockIdx.x * 256 + threadIdx.x;       // 9*8*16*512 = 589824 total
    int j    = id & 15;
    int lane = (id >> 4) & 31;
    int mt   = (id >> 9) & 15;
    int ci   = (id >> 13) & 7;
    int kk   = id >> 16;                           // 0..8
    int m  = lane & 15;
    int hi = lane >> 4;
    int v   = j >> 1;
    int pos = j & 1;
    int k = (v < 4 ? (v * 2) : (16 + (v - 4) * 2)) + pos + 8 * hi;
    int oc = mt * 16 + m;
    int c  = ci * 32 + k;
    float val = w_dcn[(oc * CIN + c) * KK + kk];
    apack[id] = (_Float16)val;
}

// ---------------------------------------------------------------------------
// Kernel 2: offset conv (f32, direct). om[b][oc][h][w], oc in 0..26.
// f32 kept deliberately: offsets feed floor(), a discontinuity.
// ---------------------------------------------------------------------------
__global__ void dcn_offset_conv(const float* __restrict__ x,
                                const float* __restrict__ w_off,
                                const float* __restrict__ b_off,
                                float* __restrict__ om) {
    int t = blockIdx.x * 256 + threadIdx.x;        // 4*27*64*64 = 442368 total
    int w  = t & 63;
    int h  = (t >> 6) & 63;
    int oc = (t >> 12) % 27;
    int b  = (t >> 12) / 27;
    float acc = b_off[oc];
    const float* xb = x + (size_t)b * CIN * HW;
    const float* wb = w_off + oc * CIN * KK;
    for (int c = 0; c < CIN; ++c) {
        const float* xc = xb + c * HW;
        const float* wc = wb + c * KK;
#pragma unroll
        for (int di = 0; di < 3; ++di) {
            int y = h + di - 1;
            if (y < 0 || y >= HH) continue;
#pragma unroll
            for (int dj = 0; dj < 3; ++dj) {
                int xx = w + dj - 1;
                if (xx < 0 || xx >= WW) continue;
                acc += xc[y * WW + xx] * wc[di * 3 + dj];
            }
        }
    }
    om[t] = acc;
}

// ---------------------------------------------------------------------------
// Kernel 3: fused bilinear-sample + WMMA GEMM.
// Grid: (W/16, H, B). Block: 256 threads = 8 wave32.
// Each WG: 16 pixels (h fixed, w0..w0+15), all 256 output channels.
// Wave wv owns M-tiles 2*wv, 2*wv+1 (32 output channels).
// K loop: 9 taps x 8 chunks of 32 input channels.
// ---------------------------------------------------------------------------
__global__ void __launch_bounds__(256)
dcn_fused_wmma(const float* __restrict__ x,
               const float* __restrict__ om,
               const _Float16* __restrict__ apack,
               float* __restrict__ out) {
    __shared__ __align__(32) _Float16 bfrag[NCHUNK * FRAG_HALVES]; // 8 KB
    __shared__ int   s_y0[KK * 16];
    __shared__ int   s_x0[KK * 16];
    __shared__ float s_ty[KK * 16];
    __shared__ float s_tx[KK * 16];
    __shared__ float s_mk[KK * 16];

    const int tid  = threadIdx.x;
    const int lane = tid & 31;         // wave32
    const int wv   = tid >> 5;         // 0..7
    const int w0 = blockIdx.x * 16;
    const int h  = blockIdx.y;
    const int b  = blockIdx.z;

    // --- precompute per-(tap,pixel) sampling state: 144 entries ---
    if (tid < KK * 16) {
        int kk = tid >> 4;
        int n  = tid & 15;
        int w  = w0 + n;
        const float* omb = om + (size_t)b * 27 * HW;
        float dy = omb[(2 * kk)     * HW + h * WW + w];
        float dx = omb[(2 * kk + 1) * HW + h * WW + w];
        float mr = omb[(18 + kk)    * HW + h * WW + w];
        float py = dy + (float)(h - 1 + kk / 3);
        float px = dx + (float)(w - 1 + kk % 3);
        float y0f = floorf(py), x0f = floorf(px);
        s_y0[tid] = (int)y0f;
        s_x0[tid] = (int)x0f;
        s_ty[tid] = py - y0f;
        s_tx[tid] = px - x0f;
        s_mk[tid] = 1.0f / (1.0f + expf(-mr));
    }

    v8f acc0 = {};
    v8f acc1 = {};

    const int n  = tid & 15;           // pixel within strip
    const int cs = tid >> 4;           // channel slice 0..15
    const float* xb = x + (size_t)b * CIN * HW;

    for (int kk = 0; kk < KK; ++kk) {
        __syncthreads();               // prev tap's WMMA reads done

        // --- cooperative bilinear sampling -> B-fragments in LDS ---
        const int idx = kk * 16 + n;
        const int y0 = s_y0[idx], x0 = s_x0[idx];
        const float ty = s_ty[idx], tx = s_tx[idx], mk = s_mk[idx];
        const int y1 = y0 + 1, x1 = x0 + 1;
        const bool vy0 = (y0 >= 0) & (y0 < HH);
        const bool vy1 = (y1 >= 0) & (y1 < HH);
        const bool vx0 = (x0 >= 0) & (x0 < WW);
        const bool vx1 = (x1 >= 0) & (x1 < WW);
        const bool v00 = vy0 & vx0, v01 = vy0 & vx1;
        const bool v10 = vy1 & vx0, v11 = vy1 & vx1;
        const float w00 = (1.f - ty) * (1.f - tx) * mk;
        const float w01 = (1.f - ty) * tx * mk;
        const float w10 = ty * (1.f - tx) * mk;
        const float w11 = ty * tx * mk;
        const int i00 = y0 * WW + x0, i01 = y0 * WW + x1;
        const int i10 = y1 * WW + x0, i11 = y1 * WW + x1;

#pragma unroll 4
        for (int i = 0; i < 16; ++i) {
            int c = (i << 4) + cs;                 // half-wave shares c -> coalesced over n
            const float* xc = xb + c * HW;
            float g00 = v00 ? xc[i00] : 0.f;
            float g01 = v01 ? xc[i01] : 0.f;
            float g10 = v10 ? xc[i10] : 0.f;
            float g11 = v11 ? xc[i11] : 0.f;
            float sv = g00 * w00 + g01 * w01 + g10 * w10 + g11 * w11;
            // ISA 16-bit B 32x16 layout: element j of lane hi*16+n holds B[(j<8 ? j : 8+j) + 8*hi][n]
            int k  = c & 31;
            int ci = c >> 5;
            int hi = (k >> 3) & 1;
            int j  = (k & 7) + ((k >> 4) << 3);
            bfrag[ci * FRAG_HALVES + (hi * 16 + n) * 16 + j] = (_Float16)sv;
        }

        __syncthreads();               // B-fragments ready

        // --- WMMA over 8 K-chunks, 2 M-tiles per wave ---
        const _Float16* apk = apack + ((size_t)kk * NCHUNK * NMT) * FRAG_HALVES;
#pragma unroll
        for (int ci = 0; ci < NCHUNK; ++ci) {
            v16h bf = *(const v16h*)(bfrag + ci * FRAG_HALVES + lane * 16);
            const _Float16* ac = apk + (size_t)ci * NMT * FRAG_HALVES;
            v16h a0 = *(const v16h*)(ac + (2 * wv)     * FRAG_HALVES + lane * 16);
            v16h a1 = *(const v16h*)(ac + (2 * wv + 1) * FRAG_HALVES + lane * 16);
            acc0 = __builtin_amdgcn_wmma_f32_16x16x32_f16(
                false, a0, false, bf, (short)0, acc0, false, false);
            acc1 = __builtin_amdgcn_wmma_f32_16x16x32_f16(
                false, a1, false, bf, (short)0, acc1, false, false);
        }
    }

    // --- epilogue: ISA C/D layout VGPR r -> M = r + 8*hi, N = lane%16 ---
    const int hi = lane >> 4;
    const int oc0 = wv * 32 + hi * 8;        // tile 2*wv
    const int oc1 = wv * 32 + 16 + hi * 8;   // tile 2*wv+1
    float* ob = out + ((size_t)b * COUT) * HW + h * WW + (w0 + n);
#pragma unroll
    for (int r = 0; r < 8; ++r) {
        ob[(size_t)(oc0 + r) * HW] = acc0[r];
        ob[(size_t)(oc1 + r) * HW] = acc1[r];
    }
}

// ---------------------------------------------------------------------------
extern "C" void kernel_launch(void* const* d_in, const int* in_sizes, int n_in,
                              void* d_out, int out_size, void* d_ws, size_t ws_size,
                              hipStream_t stream) {
    const float* x     = (const float*)d_in[0];   // (4,256,64,64)
    const float* w_off = (const float*)d_in[1];   // (27,256,3,3)
    const float* b_off = (const float*)d_in[2];   // (27,)
    const float* w_dcn = (const float*)d_in[3];   // (256,256,3,3)
    float* out = (float*)d_out;                   // (4,256,64,64)

    // workspace: om (f32, 4*27*4096) then packed f16 A-fragments (32B aligned)
    float* om = (float*)d_ws;
    const size_t om_bytes = (size_t)Bn * 27 * HW * sizeof(float);   // 1,769,472 (32B mult)
    _Float16* apack = (_Float16*)((char*)d_ws + om_bytes);

    // 1) pack weights: 9*8*16*512 = 589824 elements
    dcn_pack_wdcn<<<589824 / 256, 256, 0, stream>>>(w_dcn, apack);
    // 2) offset conv: 4*27*64*64 = 442368 outputs
    dcn_offset_conv<<<442368 / 256, 256, 0, stream>>>(x, w_off, b_off, om);
    // 3) fused sample + WMMA GEMM
    dcn_fused_wmma<<<dim3(WW / 16, HH, Bn), 256, 0, stream>>>(x, om, apack, out);
}
